// TransformerBlock_60481729462385
// MI455X (gfx1250) — matrix operand order repaired and checked by reading, not compiled, not run on hardware
//
#include <hip/hip_runtime.h>

// ---------------------------------------------------------------------------
// Types for CDNA5 WMMA (wave32): bf16 A/B fragments, f32 accumulators
// ---------------------------------------------------------------------------
typedef __attribute__((ext_vector_type(16))) __bf16 v16bf;
typedef __attribute__((ext_vector_type(8)))  __bf16 v8bf;
typedef __attribute__((ext_vector_type(8)))  float  v8f;

#define T_TOK 4096   // B*S
#define DMODEL 1024
#define NHEADS 16
#define DHEAD  64
#define SEQ    1024
#define NB     4

__device__ __forceinline__ v8f v8f_zero() {
  v8f z;
#pragma unroll
  for (int i = 0; i < 8; ++i) z[i] = 0.0f;
  return z;
}

__device__ __forceinline__ v16bf make_frag(v8bf lo, v8bf hi8) {
  v16bf f;
#pragma unroll
  for (int j = 0; j < 8; ++j) { f[j] = lo[j]; f[8 + j] = hi8[j]; }
  return f;
}

// CDNA5 async global->LDS copy (ISA §15.18.3 op 98, ASYNCcnt-tracked).
// Per-lane: LDS[vdst] = MEM[vaddr], 16 bytes.  The low 32 bits of a generic
// pointer to LDS are the wave-relative LDS byte offset (aperture mapping).
__device__ __forceinline__ void async_copy_b128(void* lds_dst, const void* gsrc) {
  unsigned lds_off = (unsigned)(size_t)lds_dst;
  unsigned long long ga = (unsigned long long)(size_t)gsrc;
  asm volatile("global_load_async_to_lds_b128 %0, %1, off"
               :: "v"(lds_off), "v"(ga)
               : "memory");
}
__device__ __forceinline__ void wait_async_le1() {
  asm volatile("s_wait_asynccnt 0x1" ::: "memory");
}
__device__ __forceinline__ void wait_async_le0() {
  asm volatile("s_wait_asynccnt 0x0" ::: "memory");
}

// ---------------------------------------------------------------------------
// Generic bf16 WMMA GEMM:  C[M,N] = A[M,K] @ W^T  (+ bias[N])
//   - A  bf16 row-major [M,K]  (activations)
//   - Bt bf16 row-major [N,K]  (weight, PRE-TRANSPOSED so each lane's
//     B-fragment -- 16 consecutive K at fixed N in the contiguous k
//     order used here -- is a contiguous 32B run, served from the 192MB L2)
//   - C f32 row-major
//   - if Cbf      != null : also store bf16 copy of C
//   - if rowScale != null : C[m,n] += rowScale[m*rsStride] * (acc + bias)
// Workgroup: 256 threads = 8 waves; block tile 64(M) x 128(N); K step 32.
// A tile staged with double-buffered ASYNC global->LDS b128 copies
// (one 16B copy per thread covers the 64x32 bf16 tile exactly).
// ---------------------------------------------------------------------------
__global__ __launch_bounds__(256)
void gemm_wmma_bf16(const __bf16* __restrict__ A, const __bf16* __restrict__ Bt,
                    const float* __restrict__ bias, float* __restrict__ C,
                    __bf16* __restrict__ Cbf, const float* __restrict__ rowScale,
                    int rsStride, int M, int N, int K)
{
  __shared__ __align__(16) __bf16 Al[2][64][32];  // 2 x 4 KB (double buffer)

  const int tid  = threadIdx.x;
  const int wave = tid >> 5;
  const int lane = tid & 31;
  const int hi   = lane >> 4;
  const int l16  = lane & 15;
  const int mi   = wave >> 1;   // 0..3 -> 16-row subtile
  const int ni   = wave & 1;    // 0..1 -> 64-col half
  const int r0   = blockIdx.y * 64;
  const int c0   = blockIdx.x * 128;

  // per-thread A-tile slice: 16 bytes
  const int arow = tid >> 2;    // 0..63
  const int aseg = tid & 3;     // 0..3
  const __bf16* abase = A + (size_t)(r0 + arow) * K + aseg * 8;

  // per-lane weight row base (column n of W = row n of Bt)
  const __bf16* brow[4];
#pragma unroll
  for (int j = 0; j < 4; ++j)
    brow[j] = Bt + (size_t)(c0 + ni * 64 + j * 16 + l16) * K + hi * 16;

  v8f acc[4];
#pragma unroll
  for (int j = 0; j < 4; ++j) acc[j] = v8f_zero();

  // prologue: async-stage A tile 0 into buffer 0
  async_copy_b128(&Al[0][arow][aseg * 8], abase);

  int buf = 0;
  for (int k0 = 0; k0 < K; k0 += 32) {
    const bool hasNext = (k0 + 32) < K;

    // issue async stage of next A tile into the other buffer
    if (hasNext)
      async_copy_b128(&Al[buf ^ 1][arow][aseg * 8], abase + k0 + 32);

    // wait for current A tile's async copy (in-order completion), then make
    // all waves' slices visible workgroup-wide
    if (hasNext) wait_async_le1(); else wait_async_le0();
    __syncthreads();

    // ---- A fragment (16x32): lane l16 = row M, K split contiguously (k = 16*hi + s)
    const int m = mi * 16 + l16;
    v16bf af = make_frag(*(const v8bf*)&Al[buf][m][hi * 16],
                         *(const v8bf*)&Al[buf][m][hi * 16 + 8]);

#pragma unroll
    for (int j = 0; j < 4; ++j) {
      const __bf16* bp = brow[j] + k0;
      if (hasNext) __builtin_prefetch(bp + 32, 0, 0);
      v16bf bfrag = make_frag(*(const v8bf*)bp, *(const v8bf*)(bp + 8));
      acc[j] = __builtin_amdgcn_wmma_f32_16x16x32_bf16(
                 false, af, false, bfrag, (short)0, acc[j], false, false);
    }
    __syncthreads();   // protect A buffer against next-next async overwrite
    buf ^= 1;
  }

  // ---- epilogue: C layout m = 8*hi + vgpr, n = l16
#pragma unroll
  for (int j = 0; j < 4; ++j) {
#pragma unroll
    for (int r = 0; r < 8; ++r) {
      const int m = r0 + mi * 16 + hi * 8 + r;
      const int n = c0 + ni * 64 + j * 16 + l16;
      float val = acc[j][r];
      if (bias) val += bias[n];
      const size_t idx = (size_t)m * N + n;
      if (rowScale) {
        C[idx] += rowScale[(size_t)m * rsStride] * val;
      } else {
        C[idx] = val;
        if (Cbf) Cbf[idx] = (__bf16)val;
      }
    }
  }
}

// ---------------------------------------------------------------------------
// Attention: one wave per (b, h, 16-query tile).  No mask (matches reference).
// scores (16 x 1024) in bf16 LDS -> softmax -> P @ V via WMMA.
// qb/kb: [B,S,H,64] bf16 (post-RoPE);  vt: [B,H,64,S] bf16;  ob: [T,1024] bf16
// ---------------------------------------------------------------------------
__global__ __launch_bounds__(32)
void attn_kernel(const __bf16* __restrict__ qb, const __bf16* __restrict__ kb,
                 const __bf16* __restrict__ vt, __bf16* __restrict__ ob)
{
  __shared__ __align__(16) __bf16 sc[16][SEQ];   // 32 KB: scores, then probs

  const int lane = threadIdx.x;
  const int hi   = lane >> 4;
  const int l16  = lane & 15;
  const int qt   = blockIdx.x;    // 0..63
  const int h    = blockIdx.y;    // 0..15
  const int b    = blockIdx.z;    // 0..3

  // Q fragments: 16 queries x 64 dims = 2 K-steps of 32
  v16bf aq[2];
  {
    const int s = qt * 16 + l16;
    const __bf16* qrow = qb + ((size_t)(b * SEQ + s) * NHEADS + h) * DHEAD;
#pragma unroll
    for (int kk = 0; kk < 2; ++kk)
      aq[kk] = make_frag(*(const v8bf*)(qrow + kk * 32 + hi * 16),
                         *(const v8bf*)(qrow + kk * 32 + hi * 16 + 8));
  }

  // ---- scores = Q @ K^T / 8
  for (int kt = 0; kt < SEQ / 16; ++kt) {
    v8f acc = v8f_zero();
    const int key = kt * 16 + l16;   // B-matrix column n = key within tile
    const __bf16* krow = kb + ((size_t)(b * SEQ + key) * NHEADS + h) * DHEAD;
#pragma unroll
    for (int kk = 0; kk < 2; ++kk) {
      v16bf bfrag = make_frag(*(const v8bf*)(krow + kk * 32 + hi * 16),
                              *(const v8bf*)(krow + kk * 32 + hi * 16 + 8));
      acc = __builtin_amdgcn_wmma_f32_16x16x32_bf16(
              false, aq[kk], false, bfrag, (short)0, acc, false, false);
    }
#pragma unroll
    for (int r = 0; r < 8; ++r)
      sc[hi * 8 + r][kt * 16 + l16] = (__bf16)(acc[r] * 0.125f);  // 1/sqrt(64)
  }
  __syncthreads();

  // ---- softmax per row: lane owns (row = l16, half = hi) of 512 columns
  {
    const int row  = l16;
    const int base = hi * 512;
    float mx = -1e30f;
    for (int c = 0; c < 512; ++c) mx = fmaxf(mx, (float)sc[row][base + c]);
    mx = fmaxf(mx, __shfl_xor(mx, 16, 32));
    float sum = 0.0f;
    for (int c = 0; c < 512; ++c) sum += __expf((float)sc[row][base + c] - mx);
    sum += __shfl_xor(sum, 16, 32);
    const float inv = 1.0f / sum;
    for (int c = 0; c < 512; ++c) {
      float p = __expf((float)sc[row][base + c] - mx) * inv;
      sc[row][base + c] = (__bf16)p;   // in-place: same owner lane
    }
  }
  __syncthreads();

  // ---- out = P(16x1024) @ V(1024x64), K step 32
  v8f oacc[4];
#pragma unroll
  for (int j = 0; j < 4; ++j) oacc[j] = v8f_zero();

  for (int kt = 0; kt < SEQ / 32; ++kt) {
    v16bf ap = make_frag(*(const v8bf*)&sc[l16][kt * 32 + hi * 16],
                         *(const v8bf*)&sc[l16][kt * 32 + hi * 16 + 8]);
#pragma unroll
    for (int j = 0; j < 4; ++j) {
      const int d = j * 16 + l16;
      const __bf16* vrow = vt + ((size_t)(b * NHEADS + h) * DHEAD + d) * SEQ + kt * 32;
      v16bf bfrag = make_frag(*(const v8bf*)(vrow + hi * 16),
                              *(const v8bf*)(vrow + hi * 16 + 8));
      oacc[j] = __builtin_amdgcn_wmma_f32_16x16x32_bf16(
                  false, ap, false, bfrag, (short)0, oacc[j], false, false);
    }
  }
#pragma unroll
  for (int j = 0; j < 4; ++j)
#pragma unroll
    for (int r = 0; r < 8; ++r) {
      const int s = qt * 16 + hi * 8 + r;
      const int d = j * 16 + l16;
      ob[(size_t)(b * SEQ + s) * DMODEL + h * DHEAD + d] = (__bf16)oacc[j][r];
    }
}

// ---------------------------------------------------------------------------
// RMSNorm: one block per token row of 1024.  eps = FLT_EPSILON (jnp.finfo).
// ---------------------------------------------------------------------------
__global__ __launch_bounds__(256)
void rmsnorm_kernel(const float* __restrict__ x, const float* __restrict__ w,
                    float* __restrict__ of, __bf16* __restrict__ ob)
{
  __shared__ float red[8];
  __shared__ float rs;
  const int t = blockIdx.x;
  const float* row = x + (size_t)t * DMODEL;
  float ss = 0.0f;
  for (int i = threadIdx.x; i < DMODEL; i += 256) { float v = row[i]; ss += v * v; }
#pragma unroll
  for (int off = 16; off > 0; off >>= 1) ss += __shfl_down(ss, off, 32);
  if ((threadIdx.x & 31) == 0) red[threadIdx.x >> 5] = ss;
  __syncthreads();
  if (threadIdx.x == 0) {
    float tot = 0.0f;
#pragma unroll
    for (int i = 0; i < 8; ++i) tot += red[i];
    rs = rsqrtf(tot / (float)DMODEL + 1.1920929e-07f);
  }
  __syncthreads();
  const float scale = rs;
  for (int i = threadIdx.x; i < DMODEL; i += 256) {
    float v = row[i] * scale * w[i];
    if (of) of[(size_t)t * DMODEL + i] = v;
    ob[(size_t)t * DMODEL + i] = (__bf16)v;
  }
}

// ---------------------------------------------------------------------------
// RoPE + pack: build bf16 q/k buffers [B,S,H,64], transposed V [B,H,64,S],
// and write post-RoPE k_rot (f32) output.  One thread per (token, head).
// ---------------------------------------------------------------------------
__global__ __launch_bounds__(256)
void pack_rope_kernel(const float* __restrict__ qbase, const float* __restrict__ qrot,
                      const float* __restrict__ kbase, const float* __restrict__ krot,
                      const float* __restrict__ vf,
                      __bf16* __restrict__ qb, __bf16* __restrict__ kb,
                      __bf16* __restrict__ vt, float* __restrict__ krot_out)
{
  const int idx = blockIdx.x * 256 + threadIdx.x;
  if (idx >= T_TOK * NHEADS) return;
  const int t = idx >> 4;
  const int h = idx & 15;
  const int b = t >> 10;
  const int s = t & 1023;

  const float* qbp = qbase + (size_t)t * 512 + h * 32;
  const float* qrp = qrot  + (size_t)t * 512 + h * 32;
  const float* kbp = kbase + (size_t)t * 512 + h * 32;
  const float* krp = krot  + (size_t)t * 512 + h * 32;
  __bf16* qdst = qb + ((size_t)t * NHEADS + h) * DHEAD;
  __bf16* kdst = kb + ((size_t)t * NHEADS + h) * DHEAD;
  float* krout = krot_out + ((size_t)t * NHEADS + h) * 32;

#pragma unroll
  for (int d = 0; d < 16; ++d) {
    const float invf = __powf(10000.0f, -((float)(2 * d) / 32.0f));
    const float fr = (float)s * invf;
    const float c = __cosf(fr), sn = __sinf(fr);
    const float q0 = qrp[d], q1 = qrp[d + 16];
    const float k0 = krp[d], k1 = krp[d + 16];
    const float qo0 = q0 * c - q1 * sn, qo1 = q1 * c + q0 * sn;
    const float ko0 = k0 * c - k1 * sn, ko1 = k1 * c + k0 * sn;
    qdst[32 + d] = (__bf16)qo0;  qdst[48 + d] = (__bf16)qo1;
    kdst[32 + d] = (__bf16)ko0;  kdst[48 + d] = (__bf16)ko1;
    krout[d] = ko0;              krout[16 + d] = ko1;
  }
#pragma unroll
  for (int d = 0; d < 32; ++d) {
    qdst[d] = (__bf16)qbp[d];
    kdst[d] = (__bf16)kbp[d];
  }
  const float* vp = vf + (size_t)t * DMODEL + h * DHEAD;
#pragma unroll
  for (int d = 0; d < DHEAD; ++d)
    vt[((size_t)(b * NHEADS + h) * DHEAD + d) * SEQ + s] = (__bf16)vp[d];
}

// ---------------------------------------------------------------------------
// Gate: logits -> softmax(8) -> top-2 -> per-expert weights (0 elsewhere).
// One wave per token.
// ---------------------------------------------------------------------------
__global__ __launch_bounds__(128)
void gate_kernel(const float* __restrict__ h2, const float* __restrict__ gw,
                 const float* __restrict__ gb, float* __restrict__ wexp)
{
  const int wave = threadIdx.x >> 5;
  const int lane = threadIdx.x & 31;
  const int t = blockIdx.x * 4 + wave;
  if (t >= T_TOK) return;
  const float* row = h2 + (size_t)t * DMODEL;
  float part[8];
#pragma unroll
  for (int e = 0; e < 8; ++e) part[e] = 0.0f;
  for (int d = lane; d < DMODEL; d += 32) {
    const float v = row[d];
    const float* g = gw + (size_t)d * 8;
#pragma unroll
    for (int e = 0; e < 8; ++e) part[e] += v * g[e];
  }
#pragma unroll
  for (int e = 0; e < 8; ++e)
#pragma unroll
    for (int off = 16; off > 0; off >>= 1) part[e] += __shfl_down(part[e], off, 32);
  if (lane == 0) {
    float pe[8];
    float mx = -1e30f;
#pragma unroll
    for (int e = 0; e < 8; ++e) { pe[e] = part[e] + gb[e]; mx = fmaxf(mx, pe[e]); }
    float sum = 0.0f;
#pragma unroll
    for (int e = 0; e < 8; ++e) { pe[e] = __expf(pe[e] - mx); sum += pe[e]; }
    const float inv = 1.0f / sum;
    int i1 = 0; float p1 = -1.0f;
#pragma unroll
    for (int e = 0; e < 8; ++e) if (pe[e] > p1) { p1 = pe[e]; i1 = e; }
    int i2 = -1; float p2 = -1.0f;
#pragma unroll
    for (int e = 0; e < 8; ++e) if (e != i1 && pe[e] > p2) { p2 = pe[e]; i2 = e; }
#pragma unroll
    for (int e = 0; e < 8; ++e)
      wexp[(size_t)t * 8 + e] = (e == i1) ? p1 * inv : ((e == i2) ? p2 * inv : 0.0f);
  }
}

// ---------------------------------------------------------------------------
// Elementwise kernels
// ---------------------------------------------------------------------------
// f32 [K,N] -> bf16 [N,K] (transposing convert; writes coalesced)
__global__ void f2bf_t_kernel(const float* __restrict__ s, __bf16* __restrict__ d,
                              int K, int N)
{
  int i = blockIdx.x * 256 + threadIdx.x;
  if (i >= N * K) return;
  const int n = i / K;
  const int k = i - n * K;
  d[i] = (__bf16)s[(size_t)k * N + n];
}

__global__ void zero_kernel(float* __restrict__ p, int n)
{ int i = blockIdx.x * 256 + threadIdx.x; if (i < n) p[i] = 0.0f; }

__global__ void add2_kernel(const float* __restrict__ a, const float* __restrict__ b,
                            float* __restrict__ o, int n)
{ int i = blockIdx.x * 256 + threadIdx.x; if (i < n) o[i] = a[i] + b[i]; }

__global__ void add3_kernel(const float* __restrict__ a, const float* __restrict__ b,
                            const float* __restrict__ c, float* __restrict__ o, int n)
{ int i = blockIdx.x * 256 + threadIdx.x; if (i < n) o[i] = a[i] + b[i] + c[i]; }

__global__ void swiglu_kernel(const float* __restrict__ g, __bf16* __restrict__ out, int n)
{
  int i = blockIdx.x * 256 + threadIdx.x;
  if (i >= n) return;
  const int row = i >> 10;           // FFN = 1024
  const int col = i & 1023;
  const float a = g[(size_t)row * 2048 + col];
  const float b = g[(size_t)row * 2048 + 1024 + col];
  const float s = a / (1.0f + __expf(-a));
  out[i] = (__bf16)(s * b);
}

// ---------------------------------------------------------------------------
// Host driver
// ---------------------------------------------------------------------------
extern "C" void kernel_launch(void* const* d_in, const int* in_sizes, int n_in,
                              void* d_out, int out_size, void* d_ws, size_t ws_size,
                              hipStream_t stream)
{
  (void)in_sizes; (void)n_in; (void)out_size; (void)ws_size;

  const float* x_in = (const float*)d_in[0];
  const float* n1w  = (const float*)d_in[1];
  const float* n2w  = (const float*)d_in[2];
  const float* b_dkv = (const float*)d_in[4];
  const float* b_dq  = (const float*)d_in[6];
  const float* b_uk  = (const float*)d_in[8];
  const float* b_uv  = (const float*)d_in[10];
  const float* b_uq  = (const float*)d_in[12];
  const float* b_qr  = (const float*)d_in[14];
  const float* b_kr  = (const float*)d_in[16];
  const float* b_wo  = (const float*)d_in[18];
  const float* gate_w = (const float*)d_in[19];
  const float* gate_b = (const float*)d_in[20];
  const float* b_s1  = (const float*)d_in[22];
  const float* b_s2  = (const float*)d_in[24];
  const float* b_r1  = (const float*)d_in[26];  // [8,2048]
  const float* b_r2  = (const float*)d_in[28];  // [8,1024]

  float* out_x    = (float*)d_out;                // [T,1024]
  float* out_ckv  = (float*)d_out + 4194304;      // [T,256]
  float* out_krot = (float*)d_out + 5242880;      // [T,16,32]

  // ---- scratch layout (bump allocator)
  char* W = (char*)d_ws;
  size_t off = 0;
  auto A = [&](size_t bytes) -> char* {
    char* p = W + off;
    off += (bytes + 255) & ~(size_t)255;
    return p;
  };

  // bf16 transposed weights [N,K]
  __bf16* w_dkv = (__bf16*)A((size_t)262144 * 2);
  __bf16* w_dq  = (__bf16*)A((size_t)262144 * 2);
  __bf16* w_uk  = (__bf16*)A((size_t)131072 * 2);
  __bf16* w_uv  = (__bf16*)A((size_t)262144 * 2);
  __bf16* w_uq  = (__bf16*)A((size_t)131072 * 2);
  __bf16* w_qr  = (__bf16*)A((size_t)131072 * 2);
  __bf16* w_kr  = (__bf16*)A((size_t)524288 * 2);
  __bf16* w_wo  = (__bf16*)A((size_t)1048576 * 2);
  __bf16* w_s1  = (__bf16*)A((size_t)2097152 * 2);
  __bf16* w_s2  = (__bf16*)A((size_t)1048576 * 2);
  __bf16* w_r1  = (__bf16*)A((size_t)16777216 * 2);
  __bf16* w_r2  = (__bf16*)A((size_t)8388608 * 2);

  // activations
  __bf16* h1b   = (__bf16*)A((size_t)T_TOK * DMODEL * 2);
  __bf16* ckvb  = (__bf16*)A((size_t)T_TOK * 256 * 2);
  __bf16* cqb   = (__bf16*)A((size_t)T_TOK * 256 * 2);
  float*  cqf   = (float*)A((size_t)T_TOK * 256 * 4);
  float*  qbase = (float*)A((size_t)T_TOK * 512 * 4);
  float*  qrotp = (float*)A((size_t)T_TOK * 512 * 4);
  float*  kbase = (float*)A((size_t)T_TOK * 512 * 4);
  float*  krotp = (float*)A((size_t)T_TOK * 512 * 4);
  float*  vf    = (float*)A((size_t)T_TOK * DMODEL * 4);
  __bf16* qb    = (__bf16*)A((size_t)T_TOK * DMODEL * 2);
  __bf16* kb    = (__bf16*)A((size_t)T_TOK * DMODEL * 2);
  __bf16* vt    = (__bf16*)A((size_t)T_TOK * DMODEL * 2);
  __bf16* ob    = (__bf16*)A((size_t)T_TOK * DMODEL * 2);
  float*  attnf = (float*)A((size_t)T_TOK * DMODEL * 4);
  float*  x1    = (float*)A((size_t)T_TOK * DMODEL * 4);
  float*  h2f   = (float*)A((size_t)T_TOK * DMODEL * 4);
  __bf16* h2b   = (__bf16*)A((size_t)T_TOK * DMODEL * 2);
  float*  gbuf  = (float*)A((size_t)T_TOK * 2048 * 4);
  __bf16* swig  = (__bf16*)A((size_t)T_TOK * 1024 * 2);
  float*  shro  = (float*)A((size_t)T_TOK * DMODEL * 4);
  float*  racc  = (float*)A((size_t)T_TOK * DMODEL * 4);
  float*  wexp  = (float*)A((size_t)T_TOK * 8 * 4);

  auto convT = [&](const float* src, __bf16* dst, int K, int N) {
    int n = N * K;
    f2bf_t_kernel<<<(n + 255) / 256, 256, 0, stream>>>(src, dst, K, N);
  };

  // ---- transposing weight conversion f32[K,N] -> bf16[N,K] (every call)
  convT((const float*)d_in[3],  w_dkv, 1024, 256);
  convT((const float*)d_in[5],  w_dq,  1024, 256);
  convT((const float*)d_in[7],  w_uk,  256,  512);
  convT((const float*)d_in[9],  w_uv,  256,  1024);
  convT((const float*)d_in[11], w_uq,  256,  512);
  convT((const float*)d_in[13], w_qr,  256,  512);
  convT((const float*)d_in[15], w_kr,  1024, 512);
  convT((const float*)d_in[17], w_wo,  1024, 1024);
  convT((const float*)d_in[21], w_s1,  1024, 2048);
  convT((const float*)d_in[23], w_s2,  1024, 1024);
  for (int e = 0; e < 8; ++e) {
    convT((const float*)d_in[25] + (size_t)e * 1024 * 2048,
          w_r1 + (size_t)e * 2048 * 1024, 1024, 2048);
    convT((const float*)d_in[27] + (size_t)e * 1024 * 1024,
          w_r2 + (size_t)e * 1024 * 1024, 1024, 1024);
  }

  auto gemm = [&](const __bf16* Am, const __bf16* Bt, const float* bias,
                  float* C, __bf16* Cbf, const float* rs, int rsStride,
                  int M, int N, int K) {
    dim3 grid(N / 128, M / 64);
    gemm_wmma_bf16<<<grid, 256, 0, stream>>>(Am, Bt, bias, C, Cbf, rs, rsStride, M, N, K);
  };

  // ---- MLA attention
  rmsnorm_kernel<<<T_TOK, 256, 0, stream>>>(x_in, n1w, (float*)nullptr, h1b);

  gemm(h1b, w_dkv, b_dkv, out_ckv, ckvb, nullptr, 0, T_TOK, 256, 1024);
  gemm(h1b, w_dq,  b_dq,  cqf,     cqb,  nullptr, 0, T_TOK, 256, 1024);
  gemm(h1b, w_kr,  b_kr,  krotp,   nullptr, nullptr, 0, T_TOK, 512, 1024);

  gemm(ckvb, w_uk, b_uk, kbase, nullptr, nullptr, 0, T_TOK, 512, 256);
  gemm(ckvb, w_uv, b_uv, vf,    nullptr, nullptr, 0, T_TOK, 1024, 256);
  gemm(cqb,  w_uq, b_uq, qbase, nullptr, nullptr, 0, T_TOK, 512, 256);
  gemm(cqb,  w_qr, b_qr, qrotp, nullptr, nullptr, 0, T_TOK, 512, 256);

  pack_rope_kernel<<<(T_TOK * NHEADS) / 256, 256, 0, stream>>>(
      qbase, qrotp, kbase, krotp, vf, qb, kb, vt, out_krot);

  attn_kernel<<<dim3(SEQ / 16, NHEADS, NB), 32, 0, stream>>>(qb, kb, vt, ob);

  gemm(ob, w_wo, b_wo, attnf, nullptr, nullptr, 0, T_TOK, 1024, 1024);
  add2_kernel<<<(T_TOK * DMODEL) / 256, 256, 0, stream>>>(x_in, attnf, x1, T_TOK * DMODEL);

  // ---- MoE
  rmsnorm_kernel<<<T_TOK, 256, 0, stream>>>(x1, n2w, h2f, h2b);
  gate_kernel<<<T_TOK / 4, 128, 0, stream>>>(h2f, gate_w, gate_b, wexp);

  // shared expert
  gemm(h2b, w_s1, b_s1, gbuf, nullptr, nullptr, 0, T_TOK, 2048, 1024);
  swiglu_kernel<<<(T_TOK * 1024) / 256, 256, 0, stream>>>(gbuf, swig, T_TOK * 1024);
  gemm(swig, w_s2, b_s2, shro, nullptr, nullptr, 0, T_TOK, 1024, 1024);

  // routed experts (dense like the reference; weight 0 contributes 0)
  zero_kernel<<<(T_TOK * DMODEL) / 256, 256, 0, stream>>>(racc, T_TOK * DMODEL);
  for (int e = 0; e < 8; ++e) {
    gemm(h2b, w_r1 + (size_t)e * 2048 * 1024, b_r1 + (size_t)e * 2048,
         gbuf, nullptr, nullptr, 0, T_TOK, 2048, 1024);
    swiglu_kernel<<<(T_TOK * 1024) / 256, 256, 0, stream>>>(gbuf, swig, T_TOK * 1024);
    gemm(swig, w_r2 + (size_t)e * 1024 * 1024, b_r2 + (size_t)e * 1024,
         racc, nullptr, wexp + e, 8, T_TOK, 1024, 1024);
  }

  add3_kernel<<<(T_TOK * DMODEL) / 256, 256, 0, stream>>>(
      x1, shro, racc, out_x, T_TOK * DMODEL);
}
